// ContextAttention_54975581388868
// MI455X (gfx1250) — matrix operand order repaired and checked by reading, not compiled, not run on hardware
//
#include <hip/hip_runtime.h>
#include <hip/hip_bf16.h>
#include <math.h>

// ---------------------------------------------------------------------------
// Problem constants (from reference): B=32 N=1025 D=768 H=12 d=64
// T=256 tiles of TS=4, K=257 context tokens.
// Padded layouts for WMMA tiling:
//   rows per batch: N 1025 -> 1040 (65 * 16), K 257 -> 288 (9 * 32)
// ---------------------------------------------------------------------------
#define BB   32
#define NN   1025
#define NP   1040
#define DD   768
#define HH   12
#define DH   64
#define TT   256
#define KK   257
#define KP   288

typedef __bf16 bf16;
typedef __attribute__((ext_vector_type(16))) __bf16 bf16x16;
typedef __attribute__((ext_vector_type(8)))  __bf16 bf16x8;
typedef __attribute__((ext_vector_type(8)))  float  f32x8;

__device__ __forceinline__ bf16 f2bf(float f) { return (bf16)f; }

__device__ __forceinline__ f32x8 wmma_bf16(bf16x16 a, bf16x16 b, f32x8 c) {
  return __builtin_amdgcn_wmma_f32_16x16x32_bf16(
      /*neg_a=*/false, a, /*neg_b=*/false, b,
      /*c_mod=*/(short)0, c, /*reuse_a=*/false, /*reuse_b=*/false);
}

// A fragment: 16x32 bf16, row-major source with row stride `ld` (elements).
// Layout used here (contiguous k, not the ISA table order): lane half h = lane>>4, M = lane&15.
//   elems [0,16) -> K = kb + 16*h + i       (contiguous 16)
__device__ __forceinline__ bf16x16 load_a_frag(const bf16* base, int ld, int kb) {
  int lane = threadIdx.x & 31;
  int half = lane >> 4;
  int m    = lane & 15;
  const bf16* r = base + (long)m * ld + kb + half * 16;
  bf16x8 lo = *(const bf16x8*)(r);
  bf16x8 hi = *(const bf16x8*)(r + 8);
  bf16x16 out;
#pragma unroll
  for (int i = 0; i < 8; ++i) { out[i] = lo[i]; out[i + 8] = hi[i]; }
  return out;
}

// B fragment for X @ W^T: B[k][n] = W[n][k]; W row-major, row stride = Kdim.
// Layout used here (matches load_a_frag): lane -> N = nbase + (lane&15); elems i -> K = kb + 16*half + i
// => one contiguous 32-byte load per lane along W's row.
__device__ __forceinline__ bf16x16 load_b_frag_wT(const bf16* W, int Kdim,
                                                  int nbase, int kb) {
  int lane = threadIdx.x & 31;
  int half = lane >> 4;
  int n    = nbase + (lane & 15);
  return *(const bf16x16*)(W + (long)n * Kdim + kb + half * 16);
}

// ---------------------------------------------------------------------------
// f32 -> bf16 converters
// ---------------------------------------------------------------------------
__global__ void cvt_flat_kernel(const float* __restrict__ in,
                                bf16* __restrict__ out, long n) {
  long i = (long)blockIdx.x * blockDim.x + threadIdx.x;
  if (i < n) out[i] = f2bf(in[i]);
}

// x (B,1025,768) f32 -> xb (B,1040,768) bf16 (pad rows left untouched)
__global__ void cvt_x_pad_kernel(const float* __restrict__ in,
                                 bf16* __restrict__ out) {
  long i = (long)blockIdx.x * blockDim.x + threadIdx.x;
  const long per = (long)NN * DD;
  if (i < (long)BB * per) {
    long b = i / per, rest = i - b * per;
    out[b * (long)NP * DD + rest] = f2bf(in[i]);
  }
}

// ---------------------------------------------------------------------------
// Tile pooling: scores = tiled . logit_w, softmax over the 4 tile elements,
// weighted sum -> ctx (B, 288-padded, 768) bf16.  ctx[b][0] = x[b][0].
// ---------------------------------------------------------------------------
__global__ __launch_bounds__(128) void ctx_kernel(const float* __restrict__ x,
                                                  const float* __restrict__ lw,
                                                  bf16* __restrict__ ctx) {
  int blk = blockIdx.x;
  int b = blk / (TT + 1);
  int t = blk % (TT + 1);
  int tid = threadIdx.x;

  if (t == TT) {  // residual token: ctx[b][0][:] = x[b][0][:]
    const float* src = x + (long)b * NN * DD;
    bf16* dst = ctx + (long)b * KP * DD;
    for (int dd = tid; dd < DD; dd += 128) dst[dd] = f2bf(src[dd]);
    return;
  }

  // tile (tr,tc) in the 16x16 grid of 2x2 tiles over the 32x32 patch grid
  int tr = t >> 4, tc = t & 15;
  long rows[4];
#pragma unroll
  for (int i = 0; i < 2; ++i)
#pragma unroll
    for (int j = 0; j < 2; ++j) {
      int p = (tr * 2 + i) * 32 + (tc * 2 + j);     // patch index
      rows[i * 2 + j] = (long)b * NN * DD + (long)(1 + p) * DD;
    }

  __shared__ float red[4][128];
  __shared__ float sc[4];
  float acc[4] = {0.f, 0.f, 0.f, 0.f};
  for (int dd = tid; dd < DD; dd += 128) {
    float w = lw[dd];
#pragma unroll
    for (int s = 0; s < 4; ++s) acc[s] += x[rows[s] + dd] * w;
  }
#pragma unroll
  for (int s = 0; s < 4; ++s) red[s][tid] = acc[s];
  __syncthreads();
  if (tid < 4) {
    float v = 0.f;
    for (int i = 0; i < 128; ++i) v += red[tid][i];
    sc[tid] = v;
  }
  __syncthreads();

  float mx = fmaxf(fmaxf(sc[0], sc[1]), fmaxf(sc[2], sc[3]));
  float e[4], sum = 0.f;
#pragma unroll
  for (int s = 0; s < 4; ++s) { e[s] = expf(sc[s] - mx); sum += e[s]; }
  float wgt[4];
#pragma unroll
  for (int s = 0; s < 4; ++s) wgt[s] = e[s] / sum;

  bf16* dst = ctx + ((long)b * KP + 1 + t) * DD;
  for (int dd = tid; dd < DD; dd += 128) {
    float v = 0.f;
#pragma unroll
    for (int s = 0; s < 4; ++s) v += wgt[s] * x[rows[s] + dd];
    dst[dd] = f2bf(v);
  }
}

// ---------------------------------------------------------------------------
// Generic WMMA GEMM: out = A @ W^T (+bias).  A (rows, 768) bf16 row-major,
// W (Nout, 768) bf16 row-major.  Rows live in per-batch padded segments.
// Block: 128 threads = 4 waves; block tile 256 rows x 64 cols;
// wave tile 64 rows x 64 cols (4 M-tiles).  The k-loop is software-pipelined
// with an explicit two-buffer ping-pong (unroll-by-2) so fragments for step
// k+1 are in flight under the 16 WMMAs of step k and no register copies are
// needed between phases.
// ---------------------------------------------------------------------------
__global__ __launch_bounds__(128) void gemm_xwT_kernel(
    const bf16* __restrict__ A, const bf16* __restrict__ W,
    int rowsPerB_in, int validPerB, int rowsPerB_out,
    int Nout, bf16* __restrict__ outB, float* __restrict__ outF,
    const float* __restrict__ bias) {
  const int Kdim = DD;
  int wave = threadIdx.x >> 5;
  int lane = threadIdx.x & 31;
  long mbase = (long)blockIdx.x * 256 + wave * 64;
  int nbase = blockIdx.y * 64;

  f32x8 c[4][4] = {};  // [m-tile][n-tile]
  const bf16* Arow = A + mbase * Kdim;

  auto loadstep = [&](int k, bf16x16 (&aa)[4], bf16x16 (&bb)[4]) {
#pragma unroll
    for (int mt = 0; mt < 4; ++mt)
      aa[mt] = load_a_frag(Arow + (long)mt * 16 * Kdim, Kdim, k);
#pragma unroll
    for (int j = 0; j < 4; ++j)
      bb[j] = load_b_frag_wT(W, Kdim, nbase + j * 16, k);
  };
  auto compute = [&](bf16x16 (&aa)[4], bf16x16 (&bb)[4]) {
#pragma unroll
    for (int j = 0; j < 4; ++j)
#pragma unroll
      for (int mt = 0; mt < 4; ++mt)
        c[mt][j] = wmma_bf16(aa[mt], bb[j], c[mt][j]);
  };

  bf16x16 aP[4], bP[4], aQ[4], bQ[4];   // ping-pong buffer sets
  loadstep(0, aP, bP);
  int kb = 0;
  for (; kb < Kdim - 64; kb += 64) {
    loadstep(kb + 32, aQ, bQ);
    if (kb + 128 < Kdim) {
      // keep the A stream two k-steps ahead (global_prefetch_b8)
#pragma unroll
      for (int mt = 0; mt < 4; ++mt)
        __builtin_prefetch(Arow + ((long)mt * 16 + (lane & 15)) * Kdim + kb + 128, 0, 3);
    }
    compute(aP, bP);                 // k-step kb
    loadstep(kb + 64, aP, bP);
    compute(aQ, bQ);                 // k-step kb+32
  }
  // kb == Kdim-64: aP/bP hold that step; one more step remains
  loadstep(kb + 32, aQ, bQ);
  compute(aP, bP);
  compute(aQ, bQ);

  int half = lane >> 4, nl = lane & 15;
#pragma unroll
  for (int mt = 0; mt < 4; ++mt)
#pragma unroll
    for (int j = 0; j < 4; ++j) {
      int n = nbase + j * 16 + nl;
#pragma unroll
      for (int r = 0; r < 8; ++r) {
        long m = mbase + mt * 16 + r + half * 8;
        long bidx = m / rowsPerB_in;
        long rr = m - bidx * rowsPerB_in;
        if (rr < validPerB) {
          long orow = bidx * (long)rowsPerB_out + rr;
          if (outB) outB[orow * (long)Nout + n] = f2bf(c[mt][j][r]);
          else      outF[orow * (long)Nout + n] = c[mt][j][r] + bias[n];
        }
      }
    }
}

// ---------------------------------------------------------------------------
// V transpose: kv (B,288,1536) v-part -> VT (B,H,64,288) so the att@V
// B-fragment becomes one contiguous 32B load (same pattern as K).
// Block handles a 32-key x 64-dd tile for one (b,h), via LDS.
// ---------------------------------------------------------------------------
__global__ __launch_bounds__(256) void vt_kernel(const bf16* __restrict__ KV,
                                                 bf16* __restrict__ VT) {
  __shared__ __align__(16) bf16 tile[32][DH + 8];
  int kt = blockIdx.x;  // key tile of 32 (0..8)
  int h  = blockIdx.y;
  int b  = blockIdx.z;
  int tid = threadIdx.x;
  {
    int key = tid >> 3;          // 0..31
    int dch = (tid & 7) * 8;     // 0..56
    const bf16* src = KV + ((long)b * KP + kt * 32 + key) * (2 * DD)
                         + DD + h * DH + dch;
    *(bf16x8*)&tile[key][dch] = *(const bf16x8*)src;
  }
  __syncthreads();
  {
    int dd  = tid >> 2;          // 0..63
    int kch = (tid & 3) * 8;     // 0..24
    bf16x8 v;
#pragma unroll
    for (int i = 0; i < 8; ++i) v[i] = tile[kch + i][dd];
    bf16* dst = VT + ((long)(b * HH + h) * DH + dd) * KP + kt * 32 + kch;
    *(bf16x8*)dst = v;
  }
}

// ---------------------------------------------------------------------------
// Attention: one wave (32 threads) per (b, h, 16-query tile).
// scores = Q Kt / 8  (36 WMMAs) -> LDS f32 -> masked softmax ->
// unnormalized exp bf16 in LDS -> att @ VT (36 WMMAs) -> normalize -> store.
// Key/V^T fragment streams are software-pipelined with explicit ping-pong.
// Q: (B,1040,768) bf16, KV: (B,288,1536) bf16, VT: (B,H,64,288) bf16.
// ---------------------------------------------------------------------------
__global__ __launch_bounds__(32) void attn_kernel(const bf16* __restrict__ Q,
                                                  const bf16* __restrict__ KV,
                                                  const bf16* __restrict__ VT,
                                                  bf16* __restrict__ O) {
  __shared__ __align__(16) float sSc[16][KP];   // raw scores (f32)
  __shared__ __align__(16) bf16  sAtt[16][KP];  // exp(s - max), unnormalized
  __shared__ float sSum[16];

  int lane = threadIdx.x;
  int qt = blockIdx.x;          // 0..64
  int h  = blockIdx.y;          // 0..11
  int b  = blockIdx.z;          // 0..31
  int half = lane >> 4, nl = lane & 15;

  const bf16* qbase = Q + ((long)b * NP + qt * 16) * DD + h * DH;
  bf16x16 a0 = load_a_frag(qbase, DD, 0);
  bf16x16 a1 = load_a_frag(qbase, DD, 32);

  // ---- scores = Q K^T over 18 key tiles, ping-pong pipelined ----
  const bf16* kbase = KV + (long)b * KP * (2 * DD) + h * DH;
  auto load_keys = [&](int nt, bf16x16& kk0, bf16x16& kk1) {
    const bf16* kr = kbase + (long)(nt * 16 + nl) * (2 * DD);
    kk0 = *(const bf16x16*)(kr + half * 16);
    kk1 = *(const bf16x16*)(kr + 32 + half * 16);
  };
  auto score_tile = [&](int nt, bf16x16 kk0, bf16x16 kk1) {
    f32x8 c = {};
    c = wmma_bf16(a0, kk0, c);
    c = wmma_bf16(a1, kk1, c);
    int key = nt * 16 + nl;
    float maskadd = (key < KK) ? 0.f : -1e30f;
#pragma unroll
    for (int r = 0; r < 8; ++r)
      sSc[r + half * 8][key] = c[r] * 0.125f + maskadd;
  };
  {
    bf16x16 p0, p1, q0, q1;
    load_keys(0, p0, p1);
    int nt = 0;
    for (; nt < KP / 16 - 2; nt += 2) {
      load_keys(nt + 1, q0, q1);
      score_tile(nt, p0, p1);
      load_keys(nt + 2, p0, p1);
      score_tile(nt + 1, q0, q1);
    }
    // nt == 16: tiles 16 and 17 remain, p holds tile 16
    load_keys(nt + 1, q0, q1);
    score_tile(nt, p0, p1);
    score_tile(nt + 1, q0, q1);
  }
  __syncthreads();

  // ---- masked softmax over 288 keys (rows handled by lanes 0..15) ----
  if (lane < 16) {
    float mx = -1e30f;
    for (int k = 0; k < KP; ++k) mx = fmaxf(mx, sSc[lane][k]);
    float sum = 0.f;
    for (int k = 0; k < KP; ++k) {
      float e = expf(sSc[lane][k] - mx);
      sum += e;
      sAtt[lane][k] = f2bf(e);
    }
    sSum[lane] = sum;
  }
  __syncthreads();

  // ---- x_attn = att @ V over 9 k-chunks, ping-pong pipelined ----
  const bf16* vtbase = VT + (long)(b * HH + h) * DH * KP;  // row dd, stride KP
#pragma unroll
  for (int dt = 0; dt < 4; ++dt) {
    const bf16* vrow = vtbase + (long)(dt * 16 + nl) * KP;
    f32x8 c = {};
    bf16x16 vP, vQ;
    vP = *(const bf16x16*)(vrow + half * 16);
    int kb = 0;
    for (; kb < KP - 64; kb += 64) {
      vQ = *(const bf16x16*)(vrow + kb + 32 + half * 16);
      {
        bf16x16 a = load_a_frag(&sAtt[0][0], KP, kb);
        c = wmma_bf16(a, vP, c);
      }
      vP = *(const bf16x16*)(vrow + kb + 64 + half * 16);
      {
        bf16x16 a = load_a_frag(&sAtt[0][0], KP, kb + 32);
        c = wmma_bf16(a, vQ, c);
      }
    }
    // kb == KP-32: one chunk left in vP
    {
      bf16x16 a = load_a_frag(&sAtt[0][0], KP, kb);
      c = wmma_bf16(a, vP, c);
    }
#pragma unroll
    for (int r = 0; r < 8; ++r) {
      int m = r + half * 8;
      float val = c[r] / sSum[m];
      O[((long)b * NP + qt * 16 + m) * DD + h * DH + dt * 16 + nl] = f2bf(val);
    }
  }
}

// ---------------------------------------------------------------------------
// Host-side orchestration
// ---------------------------------------------------------------------------
extern "C" void kernel_launch(void* const* d_in, const int* in_sizes, int n_in,
                              void* d_out, int out_size, void* d_ws, size_t ws_size,
                              hipStream_t stream) {
  const float* x    = (const float*)d_in[0];  // (32,1025,768)
  const float* lw   = (const float*)d_in[1];  // (1,768)
  const float* wq   = (const float*)d_in[2];  // (768,768)
  const float* wkv  = (const float*)d_in[3];  // (1536,768)
  const float* wo   = (const float*)d_in[4];  // (768,768)
  const float* bo   = (const float*)d_in[5];  // (768,)
  float* out = (float*)d_out;                 // (32,1025,768)

  // workspace carve-up (bf16 buffers), 256B aligned
  char* ws = (char*)d_ws;
  size_t off = 0;
  auto carve = [&](size_t bytes) -> char* {
    char* p = ws + off;
    off = (off + bytes + 255) & ~(size_t)255;
    return p;
  };
  bf16* xb   = (bf16*)carve((size_t)BB * NP * DD * 2);        // x padded
  bf16* wqb  = (bf16*)carve((size_t)DD * DD * 2);
  bf16* wkvb = (bf16*)carve((size_t)2 * DD * DD * 2);
  bf16* wob  = (bf16*)carve((size_t)DD * DD * 2);
  bf16* ctx  = (bf16*)carve((size_t)BB * KP * DD * 2);        // context padded
  bf16* kv   = (bf16*)carve((size_t)BB * KP * 2 * DD * 2);    // k|v per row
  bf16* qb   = (bf16*)carve((size_t)BB * NP * DD * 2);
  bf16* xat  = (bf16*)carve((size_t)BB * NP * DD * 2);
  bf16* vt   = (bf16*)carve((size_t)BB * HH * DH * KP * 2);   // V transposed
  (void)ws_size; (void)in_sizes; (void)n_in; (void)out_size;

  // 1) converts
  {
    long n = (long)BB * NN * DD;
    cvt_x_pad_kernel<<<(int)((n + 255) / 256), 256, 0, stream>>>(x, xb);
    long nq = (long)DD * DD;
    cvt_flat_kernel<<<(int)((nq + 255) / 256), 256, 0, stream>>>(wq, wqb, nq);
    long nkv = (long)2 * DD * DD;
    cvt_flat_kernel<<<(int)((nkv + 255) / 256), 256, 0, stream>>>(wkv, wkvb, nkv);
    cvt_flat_kernel<<<(int)((nq + 255) / 256), 256, 0, stream>>>(wo, wob, nq);
  }

  // 2) tile pooling -> ctx
  ctx_kernel<<<BB * (TT + 1), 128, 0, stream>>>(x, lw, ctx);

  // 3) q = x @ wq^T   (rows 32*1040, cols 768)
  {
    dim3 grid(BB * NP / 256, DD / 64);
    gemm_xwT_kernel<<<grid, 128, 0, stream>>>(xb, wqb, NP, NN, NP,
                                              DD, qb, nullptr, nullptr);
  }

  // 4) kv = ctx @ wkv^T  (rows 32*288, cols 1536)
  {
    dim3 grid(BB * KP / 256, (2 * DD) / 64);
    gemm_xwT_kernel<<<grid, 128, 0, stream>>>(ctx, wkvb, KP, KK, KP,
                                              2 * DD, kv, nullptr, nullptr);
  }

  // 5) transpose V for clean att@V fragments
  {
    dim3 grid(KP / 32, HH, BB);
    vt_kernel<<<grid, 256, 0, stream>>>(kv, vt);
  }

  // 6) attention
  {
    dim3 grid(NP / 16, HH, BB);
    attn_kernel<<<grid, 32, 0, stream>>>(qb, kv, vt, xat);
  }

  // 7) out = x_attn @ wo^T + b  (f32 into d_out, valid rows only)
  {
    dim3 grid(BB * NP / 256, DD / 64);
    gemm_xwT_kernel<<<grid, 128, 0, stream>>>(xat, wob, NP, NN, NN,
                                              DD, nullptr, out, bo);
  }
}
